// CausalSelfAttention_43087111913731
// MI455X (gfx1250) — matrix operand order repaired and checked by reading, not compiled, not run on hardware
//
#include <hip/hip_runtime.h>
#include <hip/hip_bf16.h>

// ---------------------------------------------------------------------------
// CausalSelfAttention for MI455X (gfx1250, wave32, WMMA).
// Pipeline: cast -> 3x GEMM (Q,K,V) -> flash-attention -> GEMM (Wo) -> out.
// Matmuls: v_wmma_f32_16x16x32_bf16 (fp32 accumulate).
// GEMM tile staging: global_load_async_to_lds_b128 + s_wait_asynccnt,
// double-buffered LDS (CDNA5 async-to-LDS path).
// ---------------------------------------------------------------------------

typedef __bf16 bf16;
typedef __bf16 v16bf __attribute__((ext_vector_type(16)));
typedef __bf16 v8bf  __attribute__((ext_vector_type(8)));
typedef __bf16 v4bf  __attribute__((ext_vector_type(4)));
typedef float  v8f   __attribute__((ext_vector_type(8)));

#define HID    2048
#define SEQ    2048
#define NBATCH 4
#define NHEAD  16
#define HDIM   128
#define MTOT   (NBATCH * SEQ)   // 8192

static __device__ __forceinline__ v8bf ld8(const bf16* p) {
    return *reinterpret_cast<const v8bf*>(p);
}
static __device__ __forceinline__ v16bf mk16(v8bf lo, v8bf hi) {
    v16bf r;
#pragma unroll
    for (int i = 0; i < 8; ++i) { r[i] = lo[i]; r[i + 8] = hi[i]; }
    return r;
}
static __device__ __forceinline__ v8f wmma_bf16(v16bf a, v16bf b, v8f c) {
    // D = A(16x32 bf16) * B(32x16 bf16) + C(16x16 f32)
    return __builtin_amdgcn_wmma_f32_16x16x32_bf16(
        /*neg_a=*/false, a, /*neg_b=*/false, b,
        /*c_mod=*/(short)0, c, /*reuse_a=*/false, /*reuse_b=*/false);
}

// CDNA5 async memory->LDS copy (tracked with ASYNCcnt).
// lds_off: LDS byte address = low 32 bits of the flat address of a __shared__
// pointer (addrspacecast local->flat is {aperture_hi, lds_offset}).
static __device__ __forceinline__ void async_g2l_b128(void* lds_dst, const void* g) {
    const uint32_t l = (uint32_t)(uintptr_t)lds_dst;
    asm volatile("global_load_async_to_lds_b128 %0, %1, off"
                 :: "v"(l), "v"((uint64_t)(uintptr_t)g)
                 : "memory");
}
template <int N>
static __device__ __forceinline__ void wait_asynccnt() {
    asm volatile("s_wait_asynccnt %0" :: "n"(N) : "memory");
}

// A-fragment (16x32, MxK): lane holds row m=lane&15; elems e<8 -> K=half*8+e,
// e>=8 -> K=16+half*8+(e-8).  base points at [row0][k0], row stride ld elems.
static __device__ __forceinline__ v16bf load_a_frag(const bf16* base, int ld) {
    const int lane = threadIdx.x & 31;
    const int m = lane & 15, half = lane >> 4;
    const bf16* p = base + (size_t)m * ld + half * 8;
    return mk16(ld8(p), ld8(p + 16));
}
// B-fragment (32x16, KxN) where B[k][n] = SRC[n][k] (SRC rows are the N dim):
// lane holds col n=lane&15; elems e<8 -> K=half*8+e, e>=8 -> K=16+half*8+(e-8).
static __device__ __forceinline__ v16bf load_b_frag(const bf16* base, int ld) {
    const int lane = threadIdx.x & 31;
    const int n = lane & 15, half = lane >> 4;
    const bf16* p = base + (size_t)n * ld + half * 8;
    return mk16(ld8(p), ld8(p + 16));
}

// ---------------------------------------------------------------------------
// fp32 -> bf16 cast, 4 elems/thread (float4 in, 8B out)
// ---------------------------------------------------------------------------
__global__ __launch_bounds__(256) void cast_f32_bf16(
    const float4* __restrict__ in, v4bf* __restrict__ out, int n4) {
    const int stride = gridDim.x * blockDim.x;
    for (int i = blockIdx.x * blockDim.x + threadIdx.x; i < n4; i += stride) {
        float4 f = in[i];
        v4bf o;
        o[0] = (bf16)f.x; o[1] = (bf16)f.y; o[2] = (bf16)f.z; o[3] = (bf16)f.w;
        out[i] = o;
    }
}

// ---------------------------------------------------------------------------
// GEMM: Y[m,n] = sum_k X[m,k] * W[n,k] + bias[n]
//   X: [MTOT, HID] bf16 row-major, W: [HID, HID] bf16 row-major (torch Linear)
// Block tile 64(M) x 128(N), BK=64; double-buffered LDS filled by async DMA.
// 8 waves, each a 32x32 sub-tile (2x2 WMMA accumulators).
// MODE 0: bf16 row-major out (Q,K)   MODE 1: bf16 out transposed [b,h,d,s] (V)
// MODE 2: f32 row-major out (final)
// ---------------------------------------------------------------------------
template <int MODE>
__global__ __launch_bounds__(256) void gemm_xwT(
    const bf16* __restrict__ X, const bf16* __restrict__ W,
    const float* __restrict__ bias, void* __restrict__ Y) {
    constexpr int BM = 64, BN = 128, BK = 64, LDT = BK + 8;  // pad: conflict-free
    __shared__ __align__(16) bf16 As[2][BM * LDT];
    __shared__ __align__(16) bf16 Bs[2][BN * LDT];

    const int bm   = blockIdx.x * BM;
    const int bn   = blockIdx.y * BN;
    const int tid  = threadIdx.x;
    const int wid  = tid >> 5;
    const int lane = tid & 31;
    const int wm   = (wid & 1) * 32;   // 2 waves along M
    const int wn   = (wid >> 1) * 32;  // 4 waves along N

    const v8f z8 = {0.f, 0.f, 0.f, 0.f, 0.f, 0.f, 0.f, 0.f};
    v8f acc[2][2] = {{z8, z8}, {z8, z8}};

    // Issue one BK-slab (A: 2 chunks/thread, B: 4 chunks/thread) as async DMA.
    auto issue_tile = [&](int k0, int buf) {
#pragma unroll
        for (int i = 0; i < 2; ++i) {
            int idx = tid + i * 256;
            int r = idx >> 3, c = (idx & 7) * 8;   // 16B chunks
            async_g2l_b128(&As[buf][r * LDT + c],
                           X + (size_t)(bm + r) * HID + k0 + c);
        }
#pragma unroll
        for (int i = 0; i < 4; ++i) {
            int idx = tid + i * 256;
            int r = idx >> 3, c = (idx & 7) * 8;
            async_g2l_b128(&Bs[buf][r * LDT + c],
                           W + (size_t)(bn + r) * HID + k0 + c);
        }
    };

    constexpr int KT = HID / BK;  // 32 K-slabs
    issue_tile(0, 0);
    for (int kt = 0; kt < KT; ++kt) {
        const int buf = kt & 1;
        if (kt + 1 < KT) {
            issue_tile((kt + 1) * BK, buf ^ 1);  // prefetch into other buffer
            wait_asynccnt<6>();                  // tile kt done; 6 newest pending
        } else {
            wait_asynccnt<0>();
        }
        __syncthreads();                         // cross-wave visibility
#pragma unroll
        for (int kk = 0; kk < BK; kk += 32) {
            v16bf a0 = load_a_frag(&As[buf][(wm + 0)  * LDT + kk], LDT);
            v16bf a1 = load_a_frag(&As[buf][(wm + 16) * LDT + kk], LDT);
            v16bf b0 = load_b_frag(&Bs[buf][(wn + 0)  * LDT + kk], LDT);
            v16bf b1 = load_b_frag(&Bs[buf][(wn + 16) * LDT + kk], LDT);
            acc[0][0] = wmma_bf16(a0, b0, acc[0][0]);
            acc[0][1] = wmma_bf16(a0, b1, acc[0][1]);
            acc[1][0] = wmma_bf16(a1, b0, acc[1][0]);
            acc[1][1] = wmma_bf16(a1, b1, acc[1][1]);
        }
        __syncthreads();                         // done reading buf before reuse
    }

    // Epilogue. C layout: lane -> col n=lane&15, reg r -> row r + 8*(lane>>4).
    const int half = lane >> 4, nn = lane & 15;
#pragma unroll
    for (int nt = 0; nt < 2; ++nt) {
        const int gn = bn + wn + nt * 16 + nn;
        const float bv = bias[gn];
#pragma unroll
        for (int mt = 0; mt < 2; ++mt) {
#pragma unroll
            for (int r = 0; r < 8; ++r) {
                const int gm = bm + wm + mt * 16 + r + half * 8;
                const float v = acc[mt][nt][r] + bv;
                if constexpr (MODE == 2) {
                    ((float*)Y)[(size_t)gm * HID + gn] = v;
                } else if constexpr (MODE == 0) {
                    ((bf16*)Y)[(size_t)gm * HID + gn] = (bf16)v;
                } else {  // V transposed: [b, h, d, s]
                    const int b = gm >> 11, s = gm & (SEQ - 1);
                    const int h = gn >> 7,  d = gn & (HDIM - 1);
                    ((bf16*)Y)[((size_t)(b * NHEAD + h) * HDIM + d) * SEQ + s] =
                        (bf16)v;
                }
            }
        }
    }
}

// ---------------------------------------------------------------------------
// Flash attention (causal).  Block = one (b,h) x 64 query rows; 4 waves x 16 rows.
// Q,K: [b, s, h*128+d] bf16.  Vt: [b, h, d, s] bf16.  Ctx out: [b, s, h*128+d].
// Per 32-key chunk: 8 WMMAs (Q.K^T over d=128) + 8 WMMAs (P.V over d=128).
// ---------------------------------------------------------------------------
__global__ __launch_bounds__(128) void attn_flash(
    const bf16* __restrict__ Q, const bf16* __restrict__ K,
    const bf16* __restrict__ Vt, bf16* __restrict__ Ctx) {
    constexpr int LDP = 40;                 // 32 + 8 pad -> conflict-free
    __shared__ bf16 Pl[4][16 * LDP];        // per-wave P transpose staging

    const int tid  = threadIdx.x;
    const int wid  = tid >> 5;
    const int lane = tid & 31;
    const int half = lane >> 4, ln = lane & 15;
    const int bh = blockIdx.y, b = bh >> 4, h = bh & 15;
    const int q0 = blockIdx.x * 64 + wid * 16;

    const bf16* qp = Q  + (size_t)b * SEQ * HID + h * HDIM;        // row stride HID
    const bf16* kp = K  + (size_t)b * SEQ * HID + h * HDIM;        // row stride HID
    const bf16* vp = Vt + (size_t)(b * NHEAD + h) * HDIM * SEQ;    // rows = d, stride SEQ
    bf16*       cp = Ctx + (size_t)b * SEQ * HID + h * HDIM;

    const float rscale = 0.08838834764831845f;  // 1/sqrt(128)
    const float LOG2E  = 1.4426950408889634f;

    // Q fragments for this wave's 16 rows: 4 x (16x32) over d = 0..127.
    v16bf qf[4];
#pragma unroll
    for (int c = 0; c < 4; ++c) {
        const bf16* p = qp + (size_t)(q0 + ln) * HID + c * 32 + half * 8;
        qf[c] = mk16(ld8(p), ld8(p + 16));
    }

    const v8f z8 = {0.f, 0.f, 0.f, 0.f, 0.f, 0.f, 0.f, 0.f};
    v8f o[8] = {z8, z8, z8, z8, z8, z8, z8, z8};  // 16 x 128 ctx accum
    float m[8], l[8];
#pragma unroll
    for (int r = 0; r < 8; ++r) { m[r] = -__builtin_inff(); l[r] = 0.f; }

    bf16* pl = Pl[wid];

    for (int j0 = 0; j0 <= q0 + 15; j0 += 32) {  // wave-uniform causal bound
        // ---- scores: two 16-key tiles, accumulate over 4 d-chunks ----
        v8f s0 = z8, s1 = z8;
#pragma unroll
        for (int c = 0; c < 4; ++c) {
            const bf16* k0p = kp + (size_t)(j0 + ln) * HID + c * 32 + half * 8;
            const bf16* k1p = kp + (size_t)(j0 + 16 + ln) * HID + c * 32 + half * 8;
            v16bf kb0 = mk16(ld8(k0p), ld8(k0p + 16));
            v16bf kb1 = mk16(ld8(k1p), ld8(k1p + 16));
            s0 = wmma_bf16(qf[c], kb0, s0);
            s1 = wmma_bf16(qf[c], kb1, s1);
        }
        // ---- mask + online softmax (row = r + 8*half, col = ln) ----
        float al[8];
#pragma unroll
        for (int r = 0; r < 8; ++r) {
            const int qrow = q0 + r + half * 8;
            float a = s0[r] * rscale;
            float c = s1[r] * rscale;
            if (j0 + ln > qrow)      a = -__builtin_inff();
            if (j0 + 16 + ln > qrow) c = -__builtin_inff();
            float rm = fmaxf(a, c);
            rm = fmaxf(rm, __shfl_xor(rm, 1, 32));
            rm = fmaxf(rm, __shfl_xor(rm, 2, 32));
            rm = fmaxf(rm, __shfl_xor(rm, 4, 32));
            rm = fmaxf(rm, __shfl_xor(rm, 8, 32));
            const float mn = fmaxf(m[r], rm);
            al[r] = exp2f((m[r] - mn) * LOG2E);
            const float p0 = exp2f((a - mn) * LOG2E);
            const float p1 = exp2f((c - mn) * LOG2E);
            float rs = p0 + p1;
            rs += __shfl_xor(rs, 1, 32);
            rs += __shfl_xor(rs, 2, 32);
            rs += __shfl_xor(rs, 4, 32);
            rs += __shfl_xor(rs, 8, 32);
            l[r] = l[r] * al[r] + rs;
            m[r] = mn;
            // stage P (C-layout) into LDS for the A-layout reload
            pl[(r + half * 8) * LDP + ln]      = (bf16)p0;
            pl[(r + half * 8) * LDP + 16 + ln] = (bf16)p1;
        }
        // rescale running context by alpha
#pragma unroll
        for (int t = 0; t < 8; ++t)
#pragma unroll
            for (int r = 0; r < 8; ++r) o[t][r] *= al[r];

        __builtin_amdgcn_wave_barrier();  // keep ds stores before the reload
        // P as A-fragment (16 q-rows x 32 keys); same-wave LDS is in-order
        const bf16* pr = pl + ln * LDP + half * 8;
        v16bf pf = mk16(*reinterpret_cast<const v8bf*>(pr),
                        *reinterpret_cast<const v8bf*>(pr + 16));
        // ---- ctx += P.V : 8 d-tiles, V transposed -> contiguous B reads ----
#pragma unroll
        for (int t = 0; t < 8; ++t) {
            const bf16* vr = vp + (size_t)(t * 16 + ln) * SEQ + j0 + half * 8;
            v16bf vf = mk16(ld8(vr), ld8(vr + 16));
            o[t] = wmma_bf16(pf, vf, o[t]);
        }
    }

    // ---- normalize and store ctx as [b, s, h*128 + d] bf16 ----
    float inv[8];
#pragma unroll
    for (int r = 0; r < 8; ++r) inv[r] = 1.0f / l[r];
#pragma unroll
    for (int t = 0; t < 8; ++t)
#pragma unroll
        for (int r = 0; r < 8; ++r)
            cp[(size_t)(q0 + r + half * 8) * HID + t * 16 + ln] =
                (bf16)(o[t][r] * inv[r]);
}

// ---------------------------------------------------------------------------
extern "C" void kernel_launch(void* const* d_in, const int* in_sizes, int n_in,
                              void* d_out, int out_size, void* d_ws, size_t ws_size,
                              hipStream_t stream) {
    (void)in_sizes; (void)n_in; (void)out_size; (void)ws_size;
    const float* x  = (const float*)d_in[0];
    // d_in[1] = mask (unused: causal mask applied analytically)
    const float* Wq = (const float*)d_in[2];
    const float* bq = (const float*)d_in[3];
    const float* Wk = (const float*)d_in[4];
    const float* bk = (const float*)d_in[5];
    const float* Wv = (const float*)d_in[6];
    const float* bv = (const float*)d_in[7];
    const float* Wo = (const float*)d_in[8];
    const float* bo = (const float*)d_in[9];
    float* out = (float*)d_out;

    // workspace carve (bf16): x, 4 weights, Q, K, Vt, Ctx  (~201 MB)
    char* ws = (char*)d_ws;
    size_t off = 0;
    auto carve = [&](size_t bytes) -> char* {
        char* p = ws + off;
        off += (bytes + 255) & ~(size_t)255;
        return p;
    };
    bf16* xb  = (bf16*)carve((size_t)MTOT * HID * 2);
    bf16* wqb = (bf16*)carve((size_t)HID * HID * 2);
    bf16* wkb = (bf16*)carve((size_t)HID * HID * 2);
    bf16* wvb = (bf16*)carve((size_t)HID * HID * 2);
    bf16* wob = (bf16*)carve((size_t)HID * HID * 2);
    bf16* qb  = (bf16*)carve((size_t)MTOT * HID * 2);
    bf16* kb  = (bf16*)carve((size_t)MTOT * HID * 2);
    bf16* vt  = (bf16*)carve((size_t)MTOT * HID * 2);
    bf16* ctx = (bf16*)carve((size_t)MTOT * HID * 2);

    cast_f32_bf16<<<2048, 256, 0, stream>>>((const float4*)x,  (v4bf*)xb,  MTOT * HID / 4);
    cast_f32_bf16<<<1024, 256, 0, stream>>>((const float4*)Wq, (v4bf*)wqb, HID * HID / 4);
    cast_f32_bf16<<<1024, 256, 0, stream>>>((const float4*)Wk, (v4bf*)wkb, HID * HID / 4);
    cast_f32_bf16<<<1024, 256, 0, stream>>>((const float4*)Wv, (v4bf*)wvb, HID * HID / 4);
    cast_f32_bf16<<<1024, 256, 0, stream>>>((const float4*)Wo, (v4bf*)wob, HID * HID / 4);

    dim3 gg(MTOT / 64, HID / 128);
    gemm_xwT<0><<<gg, 256, 0, stream>>>(xb, wqb, bq, qb);
    gemm_xwT<0><<<gg, 256, 0, stream>>>(xb, wkb, bk, kb);
    gemm_xwT<1><<<gg, 256, 0, stream>>>(xb, wvb, bv, vt);

    dim3 ga(SEQ / 64, NBATCH * NHEAD);
    attn_flash<<<ga, 128, 0, stream>>>(qb, kb, vt, ctx);

    gemm_xwT<2><<<gg, 256, 0, stream>>>(ctx, wob, bo, out);
}
